// AttentionModule_27367531610849
// MI455X (gfx1250) — compile-verified, matrix operand order checked by reading, not run on hardware
//
#include <hip/hip_runtime.h>
#include <hip/hip_bf16.h>
#include <cstdint>

// ---------- types ----------
typedef __attribute__((ext_vector_type(16))) __bf16 v16bf;
typedef __attribute__((ext_vector_type(8)))  float  v8f;

static __device__ __forceinline__ v8f wmma_bf16(v16bf a, v16bf b, v8f c) {
  // D = A(16x32) * B(32x16) + C, f32 accumulate
  return __builtin_amdgcn_wmma_f32_16x16x32_bf16(
      /*neg_a=*/false, a, /*neg_b=*/false, b,
      /*c_mod=*/(short)0, c, /*reuse_a=*/false, /*reuse_b=*/false);
}

static __device__ __forceinline__ uint16_t f2bf(float f) {
  union { float f; uint32_t u; } v; v.f = f;
  uint32_t u = v.u;
  uint32_t r = (u + 0x7FFFu + ((u >> 16) & 1u)) >> 16;
  return (uint16_t)r;
}

static __device__ __forceinline__ v16bf ldfrag(const uint16_t* p) {
  return *reinterpret_cast<const v16bf*>(p);
}

// ---------- f32 -> bf16 conversion ----------
__global__ void f32_to_bf16_kernel(const float* __restrict__ in,
                                   uint16_t* __restrict__ out, int n) {
  for (int i = blockIdx.x * blockDim.x + threadIdx.x; i < n;
       i += gridDim.x * blockDim.x)
    out[i] = f2bf(in[i]);
}

// ---------- GEMM: C[m,n] = sum_d A[m,d]*Bw[n,d] + bias[n] ----------
// A: [Mtot x 1024] bf16 row-major, Bw: [1024 x 1024] bf16 row-major.
// MODE 0: bf16 out row-major [Mtot x 1024]
// MODE 1: bf16 out transposed-per-batch [b][1024][2048]  (Mtot = b*2048 + s)
// MODE 2: f32 out row-major
template <int MODE>
__global__ __launch_bounds__(256) void gemm_nt(
    const uint16_t* __restrict__ A, const uint16_t* __restrict__ Bw,
    const float* __restrict__ bias, void* __restrict__ Out, int Mtot) {
  const int N = 1024, K = 1024;
  const int lane = threadIdx.x & 31;
  const int wave = blockIdx.x * (blockDim.x >> 5) + (threadIdx.x >> 5);
  const int ntiles = N / 32;               // 32
  const int tm = wave / ntiles, tn = wave % ntiles;
  const int m0 = tm * 32, n0 = tn * 32;
  if (m0 >= Mtot) return;                  // wave-uniform
  const int lr = lane & 15, sel = lane >> 4;

  const uint16_t* pa0 = A + (size_t)(m0 + lr) * K + 16 * sel;
  const uint16_t* pa1 = pa0 + (size_t)16 * K;
  const uint16_t* pb0 = Bw + (size_t)(n0 + lr) * K + 16 * sel;
  const uint16_t* pb1 = pb0 + (size_t)16 * K;

  v8f c00 = {}, c01 = {}, c10 = {}, c11 = {};
#pragma unroll 1
  for (int k = 0; k < K; k += 32) {
    v16bf a0 = ldfrag(pa0 + k);
    v16bf a1 = ldfrag(pa1 + k);
    v16bf b0 = ldfrag(pb0 + k);
    v16bf b1 = ldfrag(pb1 + k);
    c00 = wmma_bf16(a0, b0, c00);
    c01 = wmma_bf16(a0, b1, c01);
    c10 = wmma_bf16(a1, b0, c10);
    c11 = wmma_bf16(a1, b1, c11);
  }

  float* outF = (float*)Out;
  uint16_t* outH = (uint16_t*)Out;
#pragma unroll
  for (int r = 0; r < 8; ++r) {
    const int m_a = m0 + r + 8 * sel;
    const int m_b = m_a + 16;
    const int n_a = n0 + lr, n_b = n_a + 16;
    const float v00 = c00[r] + bias[n_a];
    const float v01 = c01[r] + bias[n_b];
    const float v10 = c10[r] + bias[n_a];
    const float v11 = c11[r] + bias[n_b];
    if (MODE == 0) {
      outH[(size_t)m_a * N + n_a] = f2bf(v00);
      outH[(size_t)m_a * N + n_b] = f2bf(v01);
      outH[(size_t)m_b * N + n_a] = f2bf(v10);
      outH[(size_t)m_b * N + n_b] = f2bf(v11);
    } else if (MODE == 1) {
      const int ba = m_a >> 11, sa = m_a & 2047;
      const int bb = m_b >> 11, sb = m_b & 2047;
      outH[((size_t)ba * 1024 + n_a) * 2048 + sa] = f2bf(v00);
      outH[((size_t)ba * 1024 + n_b) * 2048 + sa] = f2bf(v01);
      outH[((size_t)bb * 1024 + n_a) * 2048 + sb] = f2bf(v10);
      outH[((size_t)bb * 1024 + n_b) * 2048 + sb] = f2bf(v11);
    } else {
      outF[(size_t)m_a * N + n_a] = v00;
      outF[(size_t)m_a * N + n_b] = v01;
      outF[(size_t)m_b * N + n_a] = v10;
      outF[(size_t)m_b * N + n_b] = v11;
    }
  }
}

// ---------- fused attention: per (batch, 32-query block) ----------
// Q,K: [b][2048][1024] bf16 ; VT: [b][1024][2048] bf16 ; Obf: [b][2048][1024] bf16
// LDS: 32 rows x 2048 f32 scores (256KB); P repacked in-place as bf16.
__global__ __launch_bounds__(256) void attn_fused(
    const uint16_t* __restrict__ Q, const uint16_t* __restrict__ Kb,
    const uint16_t* __restrict__ VT, uint16_t* __restrict__ Obf) {
  extern __shared__ uint8_t smem_raw[];
  float* sS = (float*)smem_raw;       // [32][2048] f32
  uint16_t* sP = (uint16_t*)smem_raw; // row i halfs at offset i*4096

  const int Sdim = 2048, D = 1024;
  const int b = blockIdx.x >> 6;
  const int q0 = (blockIdx.x & 63) * 32;
  const int wave = threadIdx.x >> 5, lane = threadIdx.x & 31;
  const int lr = lane & 15, sel = lane >> 4;

  const uint16_t* Qb  = Q  + (size_t)b * Sdim * D;
  const uint16_t* Kbb = Kb + (size_t)b * Sdim * D;
  const uint16_t* VTb = VT + (size_t)b * D * Sdim;

  // ---- Phase 1: scores (scaled) into LDS. wave w owns t in [w*256, w*256+256) ----
  const int t0 = wave * 256;
  const uint16_t* pa0 = Qb + (size_t)(q0 + lr) * D + 16 * sel;
  const uint16_t* pa1 = pa0 + (size_t)16 * D;
  for (int tg = 0; tg < 4; ++tg) {      // 64 t-columns per group
    v8f c[4][2] = {};
    const uint16_t* pbb0 = Kbb + (size_t)(t0 + tg * 64 + lr) * D + 16 * sel;
    const uint16_t* pbb1 = pbb0 + (size_t)16 * D;
    const uint16_t* pbb2 = pbb0 + (size_t)32 * D;
    const uint16_t* pbb3 = pbb0 + (size_t)48 * D;
#pragma unroll 1
    for (int kc = 0; kc < D; kc += 32) {
      v16bf a0 = ldfrag(pa0 + kc);
      v16bf a1 = ldfrag(pa1 + kc);
      v16bf b0 = ldfrag(pbb0 + kc);
      v16bf b1 = ldfrag(pbb1 + kc);
      v16bf b2 = ldfrag(pbb2 + kc);
      v16bf b3 = ldfrag(pbb3 + kc);
      c[0][0] = wmma_bf16(a0, b0, c[0][0]);
      c[0][1] = wmma_bf16(a1, b0, c[0][1]);
      c[1][0] = wmma_bf16(a0, b1, c[1][0]);
      c[1][1] = wmma_bf16(a1, b1, c[1][1]);
      c[2][0] = wmma_bf16(a0, b2, c[2][0]);
      c[2][1] = wmma_bf16(a1, b2, c[2][1]);
      c[3][0] = wmma_bf16(a0, b3, c[3][0]);
      c[3][1] = wmma_bf16(a1, b3, c[3][1]);
    }
#pragma unroll
    for (int tn = 0; tn < 4; ++tn)
#pragma unroll
      for (int r = 0; r < 8; ++r) {
        const int row = r + 8 * sel;
        const int col = t0 + tg * 64 + tn * 16 + lr;
        sS[(size_t)row * Sdim + col] = c[tn][0][r] * 0.125f;
        sS[(size_t)(row + 16) * Sdim + col] = c[tn][1][r] * 0.125f;
      }
  }
  __syncthreads();

  // ---- Phase 2: softmax per row; repack P as bf16 in-place (front of row) ----
  for (int rr = 0; rr < 4; ++rr) {
    const int row = wave * 4 + rr;
    float* srow = sS + (size_t)row * Sdim;
    float mx = -3.402823466e38f;
    for (int j = lane; j < Sdim; j += 32) mx = fmaxf(mx, srow[j]);
#pragma unroll
    for (int off = 16; off; off >>= 1) mx = fmaxf(mx, __shfl_xor(mx, off, 32));
    float sum = 0.f;
    for (int j = lane; j < Sdim; j += 32) {
      float e = __expf(srow[j] - mx);
      srow[j] = e;
      sum += e;
    }
#pragma unroll
    for (int off = 16; off; off >>= 1) sum += __shfl_xor(sum, off, 32);
    const float inv = 1.0f / sum;
    uint16_t* prow = sP + (size_t)row * 4096;
    for (int j = lane; j < Sdim; j += 32) prow[j] = f2bf(srow[j] * inv);
  }
  __syncthreads();

  // ---- Phase 3: O = P(32x2048) @ V(2048x1024); wave w owns d in [w*128, +128) ----
  // Two passes of 4 n-tiles each to keep accumulator pressure at 64 VGPRs.
  uint16_t* Ob = Obf + (size_t)b * Sdim * D;
  const uint16_t* sPa0 = sP + (size_t)lr * 4096 + 16 * sel;
  const uint16_t* sPa1 = sP + (size_t)(lr + 16) * 4096 + 16 * sel;
#pragma unroll 1
  for (int half = 0; half < 2; ++half) {
    const int d0 = wave * 128 + half * 64;
    v8f acc[4][2] = {};
    const uint16_t* pv0 = VTb + (size_t)(d0 + lr) * Sdim + 16 * sel;
    const uint16_t* pv1 = pv0 + (size_t)16 * Sdim;
    const uint16_t* pv2 = pv0 + (size_t)32 * Sdim;
    const uint16_t* pv3 = pv0 + (size_t)48 * Sdim;
#pragma unroll 1
    for (int tc = 0; tc < Sdim; tc += 32) {
      v16bf a0 = *(const v16bf*)(sPa0 + tc);
      v16bf a1 = *(const v16bf*)(sPa1 + tc);
      v16bf b0 = ldfrag(pv0 + tc);
      v16bf b1 = ldfrag(pv1 + tc);
      v16bf b2 = ldfrag(pv2 + tc);
      v16bf b3 = ldfrag(pv3 + tc);
      acc[0][0] = wmma_bf16(a0, b0, acc[0][0]);
      acc[0][1] = wmma_bf16(a1, b0, acc[0][1]);
      acc[1][0] = wmma_bf16(a0, b1, acc[1][0]);
      acc[1][1] = wmma_bf16(a1, b1, acc[1][1]);
      acc[2][0] = wmma_bf16(a0, b2, acc[2][0]);
      acc[2][1] = wmma_bf16(a1, b2, acc[2][1]);
      acc[3][0] = wmma_bf16(a0, b3, acc[3][0]);
      acc[3][1] = wmma_bf16(a1, b3, acc[3][1]);
    }
#pragma unroll
    for (int nt = 0; nt < 4; ++nt)
#pragma unroll
      for (int r = 0; r < 8; ++r) {
        const int row = q0 + r + 8 * sel;
        const int col = d0 + nt * 16 + lr;
        Ob[(size_t)row * D + col] = f2bf(acc[nt][0][r]);
        Ob[(size_t)(row + 16) * D + col] = f2bf(acc[nt][1][r]);
      }
  }
}

// ---------- host-side orchestration ----------
extern "C" void kernel_launch(void* const* d_in, const int* in_sizes, int n_in,
                              void* d_out, int out_size, void* d_ws,
                              size_t ws_size, hipStream_t stream) {
  const int B = 4, S = 2048, D = 1024;
  const int M = B * S; // 8192

  const float* feat = (const float*)d_in[0];
  const float* Wq = (const float*)d_in[1];
  const float* bq = (const float*)d_in[2];
  const float* Wk = (const float*)d_in[3];
  const float* bk = (const float*)d_in[4];
  const float* Wv = (const float*)d_in[5];
  const float* bv = (const float*)d_in[6];
  const float* Wo = (const float*)d_in[7];
  const float* bo = (const float*)d_in[8];

  uint8_t* ws = (uint8_t*)d_ws;
  const size_t szX = (size_t)M * D * 2;      // 16 MB
  const size_t szW = (size_t)D * D * 2;      // 2 MB each
  uint16_t* Xbf  = (uint16_t*)(ws);                        // also reused as Abf
  uint16_t* Wqb  = (uint16_t*)(ws + szX);
  uint16_t* Wkb  = (uint16_t*)(ws + szX + szW);
  uint16_t* Wvb  = (uint16_t*)(ws + szX + 2 * szW);
  uint16_t* Wob  = (uint16_t*)(ws + szX + 3 * szW);
  uint16_t* Qbf  = (uint16_t*)(ws + szX + 4 * szW);
  uint16_t* Kbf  = (uint16_t*)(ws + szX + 4 * szW + szX);
  uint16_t* VTbf = (uint16_t*)(ws + szX + 4 * szW + 2 * szX);
  uint16_t* Abf  = Xbf; // overlay: Xbf dead after projections

  // 1) convert inputs to bf16
  f32_to_bf16_kernel<<<4096, 256, 0, stream>>>(feat, Xbf, M * D);
  f32_to_bf16_kernel<<<1024, 256, 0, stream>>>(Wq, Wqb, D * D);
  f32_to_bf16_kernel<<<1024, 256, 0, stream>>>(Wk, Wkb, D * D);
  f32_to_bf16_kernel<<<1024, 256, 0, stream>>>(Wv, Wvb, D * D);
  f32_to_bf16_kernel<<<1024, 256, 0, stream>>>(Wo, Wob, D * D);

  // 2) projections: (M/32)*(D/32) = 8192 waves -> 1024 blocks of 8 waves
  const int gemm_blocks = (M / 32) * (D / 32) / 8;
  gemm_nt<0><<<gemm_blocks, 256, 0, stream>>>(Xbf, Wqb, bq, (void*)Qbf, M);
  gemm_nt<0><<<gemm_blocks, 256, 0, stream>>>(Xbf, Wkb, bk, (void*)Kbf, M);
  gemm_nt<1><<<gemm_blocks, 256, 0, stream>>>(Xbf, Wvb, bv, (void*)VTbf, M);

  // 3) fused attention: 256 KB dynamic LDS per workgroup
  const int ldsBytes = 32 * 2048 * 4;
  hipFuncSetAttribute((const void*)attn_fused,
                      hipFuncAttributeMaxDynamicSharedMemorySize, ldsBytes);
  attn_fused<<<B * (S / 32), 256, ldsBytes, stream>>>(Qbf, Kbf, VTbf, Abf);

  // 4) output projection -> f32 d_out
  gemm_nt<2><<<gemm_blocks, 256, 0, stream>>>(Abf, Wob, bo, d_out, M);
}
